// MixtureOfDepth_27015344292001
// MI455X (gfx1250) — compile-verified, matrix operand order checked by reading, not run on hardware
//
#include <hip/hip_runtime.h>
#include <stdint.h>

// ---------------------------------------------------------------------------
// MixtureOfDepth forward for gfx1250 (MI455X): bf16 WMMA GEMM pipeline with
// async global->LDS staging and double-buffered tiles.
// B=2, S=4096, D=1024, H=16, HD=64, DFF=4096, k=512, m=511 (padded to 512).
// ---------------------------------------------------------------------------

typedef __attribute__((ext_vector_type(16))) __bf16 bf16x16;
typedef __attribute__((ext_vector_type(8)))  __bf16 bf16x8;
typedef __attribute__((ext_vector_type(8)))  float  f32x8;

#define NEGBIG (-1.0e9f)

__device__ __forceinline__ unsigned short f2bf(float f) {
  unsigned int u = __float_as_uint(f);
  unsigned int r = u + 0x7fffu + ((u >> 16) & 1u); // round-to-nearest-even
  return (unsigned short)(r >> 16);
}

__device__ __forceinline__ float gelu_tanh(float x) {
  float x3 = x * x * x;
  return 0.5f * x * (1.0f + tanhf(0.79788456080286535588f * (x + 0.044715f * x3)));
}

// Low 32 bits of a generic pointer to LDS == LDS byte offset.
__device__ __forceinline__ unsigned lds_off(const void* p) {
  return (unsigned)(uintptr_t)p;
}

// Async contiguous 16-byte global->LDS copy (GV mode, per-lane address).
__device__ __forceinline__ void async_copy_b128(unsigned lds, const void* gp) {
  unsigned long long ga = (unsigned long long)(uintptr_t)gp;
  asm volatile("global_load_async_to_lds_b128 %0, %1, off"
               :: "v"(lds), "v"(ga) : "memory");
}
__device__ __forceinline__ void wait_async0() {
  asm volatile("s_wait_asynccnt 0x0" ::: "memory");
}

// --------------------------- router logits ---------------------------------
__global__ __launch_bounds__(256) void router_kernel(
    const float* __restrict__ hs, const float* __restrict__ rwv,
    float* __restrict__ wts) {
  int tid = threadIdx.x, wave = tid >> 5, lane = tid & 31;
  int token = blockIdx.x * 8 + wave;          // 0..B*S-1
  const float* p = hs + (size_t)token * 1024;
  float acc = 0.0f;
#pragma unroll 8
  for (int e = 0; e < 32; ++e) acc += p[lane + 32 * e] * rwv[lane + 32 * e];
#pragma unroll
  for (int o = 16; o > 0; o >>= 1) acc += __shfl_xor(acc, o, 32);
  if (lane == 0) wts[token] = acc;
}

// ----------------------------- top-k (per batch) ---------------------------
__global__ __launch_bounds__(1024) void topk_kernel(
    const float* __restrict__ wts, int* __restrict__ sel_idx,
    float* __restrict__ rw, float* __restrict__ thr) {
  __shared__ unsigned long long key[4096];
  __shared__ int sidx[512];
  int b = blockIdx.x, tid = threadIdx.x;
  const float* w = wts + (size_t)b * 4096;
  for (int i = tid; i < 4096; i += 1024) {
    unsigned int u = __float_as_uint(w[i]);
    unsigned int a = (u >> 31) ? ~u : (u | 0x80000000u); // ascending-sortable
    unsigned int d = ~a;                                  // descending by value
    key[i] = ((unsigned long long)d << 32) | (unsigned int)i;
  }
  for (int k = 2; k <= 4096; k <<= 1) {
    for (int j = k >> 1; j > 0; j >>= 1) {
      __syncthreads();
      for (int i = tid; i < 4096; i += 1024) {
        int ixj = i ^ j;
        if (ixj > i) {
          unsigned long long a = key[i], c = key[ixj];
          bool up = ((i & k) == 0);
          if (up ? (a > c) : (a < c)) { key[i] = c; key[ixj] = a; }
        }
      }
    }
  }
  __syncthreads();
  if (tid == 0) {
    int tix = (int)(unsigned int)key[511]; // k-th (512th) largest
    thr[b] = w[tix];
  }
  if (tid < 512) sidx[tid] = (tid < 511) ? (int)(unsigned int)key[tid] : 0x7FFFFFFF;
  for (int k = 2; k <= 512; k <<= 1) {
    for (int j = k >> 1; j > 0; j >>= 1) {
      __syncthreads();
      if (tid < 512) {
        int i = tid, ixj = i ^ j;
        if (ixj > i && ixj < 512) {
          int a = sidx[i], c = sidx[ixj];
          bool up = ((i & k) == 0);
          if (up ? (a > c) : (a < c)) { sidx[i] = c; sidx[ixj] = a; }
        }
      }
    }
  }
  __syncthreads();
  if (tid < 512) {
    if (tid < 511) {
      int s = sidx[tid];
      sel_idx[b * 512 + tid] = s;
      rw[b * 512 + tid] = w[s];
    } else {
      sel_idx[b * 512 + tid] = 0;   // safe padding
      rw[b * 512 + tid] = 0.0f;
    }
  }
}

// ---------------------- masked pass-through init of out --------------------
__global__ __launch_bounds__(256) void init_out_kernel(
    const float* __restrict__ hs, const float* __restrict__ wts,
    const float* __restrict__ thr, float* __restrict__ out) {
  int r = blockIdx.x;                 // 0..B*S-1
  int b = r >> 12;                    // S=4096
  bool zero = wts[r] > thr[b];
  size_t base = (size_t)r * 1024 + threadIdx.x * 4;
  float4 v;
  if (zero) v = make_float4(0.f, 0.f, 0.f, 0.f);
  else      v = *reinterpret_cast<const float4*>(hs + base);
  *reinterpret_cast<float4*>(out + base) = v;
}

// ------------------- gather (optional) + LayerNorm -> bf16 -----------------
__global__ __launch_bounds__(256) void gather_ln_kernel(
    const float* __restrict__ src, const int* __restrict__ sel,
    const float* __restrict__ g, const float* __restrict__ beta,
    float* __restrict__ xout, unsigned short* __restrict__ hout, int S_src) {
  __shared__ float ssum[256], ssq[256];
  int r = blockIdx.x;                 // 0..B*512-1
  int b = r >> 9, i = r & 511;
  int tid = threadIdx.x;
  bool zero = false;
  const float* row;
  if (sel) {
    if (i == 511) zero = true;
    int s = sel[b * 512 + i];
    row = src + ((size_t)b * S_src + s) * 1024;
  } else {
    row = src + (size_t)r * 1024;
  }
  float4 v = zero ? make_float4(0.f, 0.f, 0.f, 0.f)
                  : *reinterpret_cast<const float4*>(row + tid * 4);
  ssum[tid] = v.x + v.y + v.z + v.w;
  ssq[tid]  = v.x * v.x + v.y * v.y + v.z * v.z + v.w * v.w;
  __syncthreads();
  for (int o = 128; o > 0; o >>= 1) {
    if (tid < o) { ssum[tid] += ssum[tid + o]; ssq[tid] += ssq[tid + o]; }
    __syncthreads();
  }
  float mean = ssum[0] * (1.0f / 1024.0f);
  float var  = ssq[0] * (1.0f / 1024.0f) - mean * mean;
  float rs   = rsqrtf(var + 1e-5f);
  size_t obase = (size_t)r * 1024 + tid * 4;
  float e[4] = {v.x, v.y, v.z, v.w};
#pragma unroll
  for (int j = 0; j < 4; ++j) {
    int c = tid * 4 + j;
    float h = zero ? 0.0f : (e[j] - mean) * rs * g[c] + beta[c];
    hout[obase + j] = f2bf(h);
    if (xout) xout[obase + j] = e[j];
  }
}

// ------------------------------ f32 -> bf16 --------------------------------
__global__ __launch_bounds__(256) void conv_f32_bf16(
    const float* __restrict__ in, unsigned short* __restrict__ out, int n) {
  int idx = (blockIdx.x * 256 + threadIdx.x) * 4;
  if (idx >= n) return;
  float4 v = *reinterpret_cast<const float4*>(in + idx);
  ushort4 o = make_ushort4(f2bf(v.x), f2bf(v.y), f2bf(v.z), f2bf(v.w));
  *reinterpret_cast<ushort4*>(out + idx) = o;
}

// --------------------------- gelu(f32) -> bf16 -----------------------------
__global__ __launch_bounds__(256) void gelu_bf16_kernel(
    const float* __restrict__ in, unsigned short* __restrict__ out, int n) {
  int idx = (blockIdx.x * 256 + threadIdx.x) * 4;
  if (idx >= n) return;
  float4 v = *reinterpret_cast<const float4*>(in + idx);
  ushort4 o = make_ushort4(f2bf(gelu_tanh(v.x)), f2bf(gelu_tanh(v.y)),
                           f2bf(gelu_tanh(v.z)), f2bf(gelu_tanh(v.w)));
  *reinterpret_cast<ushort4*>(out + idx) = o;
}

// ------------------------------- RoPE --------------------------------------
__global__ __launch_bounds__(256) void rope_kernel(
    const float* __restrict__ qf, const float* __restrict__ kf,
    const int* __restrict__ sel, const int* __restrict__ pids,
    unsigned short* __restrict__ qb, unsigned short* __restrict__ kb) {
  int r = blockIdx.x;                 // 0..B*512-1
  int b = r >> 9, i = r & 511;
  float p = (float)pids[sel[b * 512 + i]];
  int tid = threadIdx.x, grp = tid >> 5, d = tid & 31;
  float inv = powf(10000.0f, -(float)d * (1.0f / 32.0f));
  float ang = p * inv, c = cosf(ang), s = sinf(ang);
  for (int h = grp; h < 16; h += 8) {
    size_t base = ((size_t)r * 16 + h) * 64;
    float q1 = qf[base + d], q2 = qf[base + d + 32];
    qb[base + d]      = f2bf(q1 * c - q2 * s);
    qb[base + d + 32] = f2bf(q1 * s + q2 * c);
    float k1 = kf[base + d], k2 = kf[base + d + 32];
    kb[base + d]      = f2bf(k1 * c - k2 * s);
    kb[base + d + 32] = f2bf(k1 * s + k2 * c);
  }
}

// -------------------------- bf16 WMMA GEMM ---------------------------------
// C[M,N](f32, += addsrc) = A[M,K](bf16) * B[K,N](bf16)  (TRANSB: B is [N,K]).
// Block tile 64x64, K-step 32, 8 waves -> each wave 16x32 via two
// v_wmma_f32_16x16x32_bf16.  LDS tiles are double-buffered and filled with
// GLOBAL_LOAD_ASYNC_TO_LDS_B128 (ASYNCcnt); B is kept n-major (B^T) in LDS so
// every fragment is two contiguous ds_load_b128 per lane (ISA 7.12.2 layout).
template <bool TRANSB>
__global__ __launch_bounds__(256) void gemm_bf16(
    const unsigned short* __restrict__ A, const unsigned short* __restrict__ Bm,
    float* __restrict__ C, const float* __restrict__ addsrc,
    int M, int N, int K, int lda, int ldb, int ldc, int zdiv,
    long long sAo, long long sAi, long long sBo, long long sBi,
    long long sCo, long long sCi) {
  __shared__ unsigned short As[2][64][40];   // m-major, k contiguous (+8 pad)
  __shared__ unsigned short Bt[2][64][40];   // n-major, k contiguous (+8 pad)
  int z = blockIdx.z;
  int zo = z / zdiv, zi = z % zdiv;
  const unsigned short* Ap = A + (size_t)zo * sAo + (size_t)zi * sAi;
  const unsigned short* Bp = Bm + (size_t)zo * sBo + (size_t)zi * sBi;
  size_t offC = (size_t)zo * sCo + (size_t)zi * sCi;
  float* Cp = C + offC;
  const float* Sp = addsrc ? addsrc + offC : nullptr;

  int m0 = blockIdx.y * 64, n0 = blockIdx.x * 64;
  int tid = threadIdx.x, lane = tid & 31, wave = tid >> 5;
  int wm = wave >> 1, wn = wave & 1;

  f32x8 acc0 = {}, acc1 = {};
  int row_a = tid >> 2, seg_a = tid & 3;           // 64 rows x 4 segs of 8

  auto stage = [&](int k0, int buf) {
    // A tile 64x32: contiguous global -> contiguous LDS, async B128.
    async_copy_b128(lds_off(&As[buf][row_a][seg_a * 8]),
                    Ap + (size_t)(m0 + row_a) * lda + k0 + seg_a * 8);
    if (TRANSB) {
      // B is [N,K] row-major: rows are k-contiguous -> async B128 into Bt.
      async_copy_b128(lds_off(&Bt[buf][row_a][seg_a * 8]),
                      Bp + (size_t)(n0 + row_a) * ldb + k0 + seg_a * 8);
    } else {
      // B is [K,N] row-major: transpose while staging (8 x ds_store_b16).
      int kr = tid >> 3, seg = tid & 7;
      const unsigned short* src = Bp + (size_t)(k0 + kr) * ldb + n0 + seg * 8;
      unsigned short tmp[8];
      *reinterpret_cast<uint4*>(tmp) = *reinterpret_cast<const uint4*>(src);
#pragma unroll
      for (int j = 0; j < 8; ++j) Bt[buf][seg * 8 + j][kr] = tmp[j];
    }
  };

  stage(0, 0);
  int nk = K >> 5;
  int arow = wm * 16 + (lane & 15);
  int kbase = (lane >> 4) * 8;
  int bc = wn * 32 + (lane & 15);
#pragma unroll 2
  for (int kk = 0; kk < nk; ++kk) {
    wait_async0();               // own tile-kk async copies have landed
    __syncthreads();             // everyone's copies visible; prev reads done
    if (kk + 1 < nk) stage((kk + 1) << 5, (kk + 1) & 1);
    int buf = kk & 1;

    bf16x8 alo = *reinterpret_cast<const bf16x8*>(&As[buf][arow][kbase]);
    bf16x8 ahi = *reinterpret_cast<const bf16x8*>(&As[buf][arow][kbase + 16]);
    bf16x16 a = __builtin_shufflevector(alo, ahi, 0, 1, 2, 3, 4, 5, 6, 7,
                                        8, 9, 10, 11, 12, 13, 14, 15);
    bf16x8 b0lo = *reinterpret_cast<const bf16x8*>(&Bt[buf][bc][kbase]);
    bf16x8 b0hi = *reinterpret_cast<const bf16x8*>(&Bt[buf][bc][kbase + 16]);
    bf16x8 b1lo = *reinterpret_cast<const bf16x8*>(&Bt[buf][bc + 16][kbase]);
    bf16x8 b1hi = *reinterpret_cast<const bf16x8*>(&Bt[buf][bc + 16][kbase + 16]);
    bf16x16 b0 = __builtin_shufflevector(b0lo, b0hi, 0, 1, 2, 3, 4, 5, 6, 7,
                                         8, 9, 10, 11, 12, 13, 14, 15);
    bf16x16 b1 = __builtin_shufflevector(b1lo, b1hi, 0, 1, 2, 3, 4, 5, 6, 7,
                                         8, 9, 10, 11, 12, 13, 14, 15);
    acc0 = __builtin_amdgcn_wmma_f32_16x16x32_bf16(false, a, false, b0,
                                                   (short)0, acc0, false, false);
    acc1 = __builtin_amdgcn_wmma_f32_16x16x32_bf16(false, a, false, b1,
                                                   (short)0, acc1, false, false);
  }

  // epilogue: C/D layout: row = (lane>>4)*8 + vr, col = lane&15 (ISA 7.12.2)
  int crow = m0 + wm * 16 + (lane >> 4) * 8;
  int ccol = n0 + wn * 32 + (lane & 15);
  if (Sp) {
#pragma unroll
    for (int vr = 0; vr < 8; ++vr) {
      size_t idx = (size_t)(crow + vr) * ldc + ccol;
      Cp[idx]      = acc0[vr] + Sp[idx];
      Cp[idx + 16] = acc1[vr] + Sp[idx + 16];
    }
  } else {
#pragma unroll
    for (int vr = 0; vr < 8; ++vr) {
      size_t idx = (size_t)(crow + vr) * ldc + ccol;
      Cp[idx]      = acc0[vr];
      Cp[idx + 16] = acc1[vr];
    }
  }
}

// --------------------------- masked softmax --------------------------------
__global__ __launch_bounds__(256) void softmax_kernel(
    const float* __restrict__ scores, unsigned short* __restrict__ attnb) {
  __shared__ float red[256];
  int row = blockIdx.x;               // 0..B*H*512-1
  int i = row & 511;
  const float* p = scores + (size_t)row * 512;
  int tid = threadIdx.x;
  const float scale = 0.125f;         // 1/sqrt(64)
  int j0 = tid, j1 = tid + 256;
  bool v0 = (j0 <= i) && (j0 < 511);
  bool v1 = (j1 <= i) && (j1 < 511);
  float s0 = v0 ? p[j0] * scale : NEGBIG;
  float s1 = v1 ? p[j1] * scale : NEGBIG;
  red[tid] = fmaxf(s0, s1);
  __syncthreads();
  for (int o = 128; o > 0; o >>= 1) {
    if (tid < o) red[tid] = fmaxf(red[tid], red[tid + o]);
    __syncthreads();
  }
  float mx = red[0];
  __syncthreads();
  float e0 = v0 ? expf(s0 - mx) : 0.0f;
  float e1 = v1 ? expf(s1 - mx) : 0.0f;
  red[tid] = e0 + e1;
  __syncthreads();
  for (int o = 128; o > 0; o >>= 1) {
    if (tid < o) red[tid] += red[tid + o];
    __syncthreads();
  }
  float inv = 1.0f / red[0];
  unsigned short* q = attnb + (size_t)row * 512;
  q[j0] = f2bf(e0 * inv);
  q[j1] = f2bf(e1 * inv);
}

// ------------------------------- scatter -----------------------------------
__global__ __launch_bounds__(256) void scatter_kernel(
    const float* __restrict__ x3, const int* __restrict__ sel,
    const float* __restrict__ rw, float* __restrict__ out) {
  int r = blockIdx.x;                 // 0..B*512-1
  int b = r >> 9, i = r & 511;
  if (i == 511) return;               // padded row
  int s = sel[b * 512 + i];
  float w = rw[b * 512 + i];
  int tid = threadIdx.x;
  float4 v = *reinterpret_cast<const float4*>(x3 + (size_t)r * 1024 + tid * 4);
  v.x *= w; v.y *= w; v.z *= w; v.w *= w;
  *reinterpret_cast<float4*>(out + ((size_t)b * 4096 + s) * 1024 + tid * 4) = v;
}

// ===========================================================================
extern "C" void kernel_launch(void* const* d_in, const int* in_sizes, int n_in,
                              void* d_out, int out_size, void* d_ws, size_t ws_size,
                              hipStream_t stream) {
  (void)in_sizes; (void)n_in; (void)out_size; (void)ws_size;
  const float* hidden   = (const float*)d_in[0];
  // d_in[1] attention_mask: pure causal; recomputed from sorted indices.
  const int*   pids     = (const int*)d_in[2];
  const float* router_w = (const float*)d_in[3];
  const float* Wq = (const float*)d_in[4];
  const float* Wk = (const float*)d_in[5];
  const float* Wv = (const float*)d_in[6];
  const float* Wo = (const float*)d_in[7];
  const float* W1 = (const float*)d_in[8];
  const float* W2 = (const float*)d_in[9];
  const float* ln1_g = (const float*)d_in[10];
  const float* ln1_b = (const float*)d_in[11];
  const float* ln2_g = (const float*)d_in[12];
  const float* ln2_b = (const float*)d_in[13];
  float* out = (float*)d_out;

  const int Bc = 2, Sc = 4096, Dc = 1024, Hc = 16, DFF = 4096, Mp = 512;
  const long long MD  = (long long)Bc * Mp * Dc;     // 1,048,576
  const long long MDF = (long long)Bc * Mp * DFF;    // 4,194,304
  const long long SQ  = (long long)Mp * Mp;          // 262,144

  char* base = (char*)d_ws;
  size_t off = 0;
  auto alloc = [&](size_t bytes) -> char* {
    char* p = base + off;
    off = (off + bytes + 255) & ~(size_t)255;
    return p;
  };
  float*          wts   = (float*)alloc((size_t)Bc * Sc * 4);
  float*          thr   = (float*)alloc(Bc * 4);
  int*            sel   = (int*)alloc((size_t)Bc * Mp * 4);
  float*          rwbuf = (float*)alloc((size_t)Bc * Mp * 4);
  float*          x     = (float*)alloc(MD * 4);
  unsigned short* hbf   = (unsigned short*)alloc(MD * 2);
  float*          qf    = (float*)alloc(MD * 4);
  float*          kf    = (float*)alloc(MD * 4);
  float*          vf    = (float*)alloc(MD * 4);
  unsigned short* qb    = (unsigned short*)alloc(MD * 2);
  unsigned short* kb    = (unsigned short*)alloc(MD * 2);
  unsigned short* vb    = (unsigned short*)alloc(MD * 2);
  float*          sco   = (float*)alloc((size_t)Bc * Hc * SQ * 4);
  unsigned short* attnb = (unsigned short*)alloc((size_t)Bc * Hc * SQ * 2);
  float*          ctxf  = (float*)alloc(MD * 4);
  unsigned short* ctxb  = (unsigned short*)alloc(MD * 2);
  float*          xa    = (float*)alloc(MD * 4);
  unsigned short* h2b   = (unsigned short*)alloc(MD * 2);
  float*          ff    = (float*)alloc(MDF * 4);
  unsigned short* gb    = (unsigned short*)alloc(MDF * 2);
  float*          x3    = (float*)alloc(MD * 4);
  unsigned short* Wqb   = (unsigned short*)alloc((size_t)Dc * Dc * 2);
  unsigned short* Wkb   = (unsigned short*)alloc((size_t)Dc * Dc * 2);
  unsigned short* Wvb   = (unsigned short*)alloc((size_t)Dc * Dc * 2);
  unsigned short* Wob   = (unsigned short*)alloc((size_t)Dc * Dc * 2);
  unsigned short* W1b   = (unsigned short*)alloc((size_t)Dc * DFF * 2);
  unsigned short* W2b   = (unsigned short*)alloc((size_t)DFF * Dc * 2);

  auto conv = [&](const float* in, unsigned short* o, int n) {
    conv_f32_bf16<<<dim3(n / 1024), dim3(256), 0, stream>>>(in, o, n);
  };
  auto gemm = [&](bool transb, const unsigned short* A, const unsigned short* Bm,
                  float* C, const float* add, int M, int N, int K,
                  int lda, int ldb, int ldc, int nz, int zdiv,
                  long long sAo, long long sAi, long long sBo, long long sBi,
                  long long sCo, long long sCi) {
    dim3 g(N / 64, M / 64, nz), blk(256);
    if (transb)
      gemm_bf16<true><<<g, blk, 0, stream>>>(A, Bm, C, add, M, N, K, lda, ldb,
                                             ldc, zdiv, sAo, sAi, sBo, sBi, sCo, sCi);
    else
      gemm_bf16<false><<<g, blk, 0, stream>>>(A, Bm, C, add, M, N, K, lda, ldb,
                                              ldc, zdiv, sAo, sAi, sBo, sBi, sCo, sCi);
  };

  // 1) router + top-k + passthrough init
  router_kernel<<<dim3(Bc * Sc / 8), dim3(256), 0, stream>>>(hidden, router_w, wts);
  topk_kernel<<<dim3(Bc), dim3(1024), 0, stream>>>(wts, sel, rwbuf, thr);
  init_out_kernel<<<dim3(Bc * Sc), dim3(256), 0, stream>>>(hidden, wts, thr, out);

  // 2) weights -> bf16
  conv(Wq, Wqb, Dc * Dc); conv(Wk, Wkb, Dc * Dc);
  conv(Wv, Wvb, Dc * Dc); conv(Wo, Wob, Dc * Dc);
  conv(W1, W1b, Dc * DFF); conv(W2, W2b, DFF * Dc);

  // 3) gather + LN1 -> x (f32), hbf (bf16)
  gather_ln_kernel<<<dim3(Bc * Mp), dim3(256), 0, stream>>>(
      hidden, sel, ln1_g, ln1_b, x, hbf, Sc);

  // 4) QKV GEMMs (batches folded: M = B*512)
  gemm(false, hbf, Wqb, qf, nullptr, Bc * Mp, Dc, Dc, Dc, Dc, Dc, 1, 1, 0,0,0,0,0,0);
  gemm(false, hbf, Wkb, kf, nullptr, Bc * Mp, Dc, Dc, Dc, Dc, Dc, 1, 1, 0,0,0,0,0,0);
  gemm(false, hbf, Wvb, vf, nullptr, Bc * Mp, Dc, Dc, Dc, Dc, Dc, 1, 1, 0,0,0,0,0,0);

  // 5) RoPE -> qb,kb (bf16); v -> bf16
  rope_kernel<<<dim3(Bc * Mp), dim3(256), 0, stream>>>(qf, kf, sel, pids, qb, kb);
  conv(vf, vb, (int)MD);

  // 6) scores = q @ k^T per (b,h):  z = b*16+h, zdiv=16  (TRANSB path -> async)
  gemm(true, qb, kb, sco, nullptr, Mp, Mp, 64, Dc, Dc, Mp, Bc * Hc, Hc,
       (long long)Mp * Dc, 64, (long long)Mp * Dc, 64, (long long)Hc * SQ, SQ);

  // 7) masked softmax -> bf16 attn
  softmax_kernel<<<dim3(Bc * Hc * Mp), dim3(256), 0, stream>>>(sco, attnb);

  // 8) ctx = attn @ v per (b,h)
  gemm(false, attnb, vb, ctxf, nullptr, Mp, 64, Mp, Mp, Dc, Dc, Bc * Hc, Hc,
       (long long)Hc * SQ, SQ, (long long)Mp * Dc, 64, (long long)Mp * Dc, 64);
  conv(ctxf, ctxb, (int)MD);

  // 9) xa = x + ctx @ Wo
  gemm(false, ctxb, Wob, xa, x, Bc * Mp, Dc, Dc, Dc, Dc, Dc, 1, 1, 0,0,0,0,0,0);

  // 10) LN2 -> h2b (bf16)
  gather_ln_kernel<<<dim3(Bc * Mp), dim3(256), 0, stream>>>(
      xa, nullptr, ln2_g, ln2_b, nullptr, h2b, Sc);

  // 11) MLP: ff = h2 @ W1 ; g = gelu(ff) bf16 ; x3 = xa + g @ W2
  gemm(false, h2b, W1b, ff, nullptr, Bc * Mp, DFF, Dc, Dc, DFF, DFF, 1, 1, 0,0,0,0,0,0);
  gelu_bf16_kernel<<<dim3((int)(MDF / 1024)), dim3(256), 0, stream>>>(ff, gb, (int)MDF);
  gemm(false, gb, W2b, x3, xa, Bc * Mp, Dc, DFF, DFF, Dc, Dc, 1, 1, 0,0,0,0,0,0);

  // 12) scatter scaled results
  scatter_kernel<<<dim3(Bc * Mp), dim3(256), 0, stream>>>(x3, sel, rwbuf, out);
}
